// NeuralAnalyticsModel_65025804861855
// MI455X (gfx1250) — matrix operand order repaired and checked by reading, not compiled, not run on hardware
//
#include <hip/hip_runtime.h>
#include <hip/hip_bf16.h>

typedef _Float16 half8 __attribute__((ext_vector_type(8)));
typedef _Float16 v16h  __attribute__((ext_vector_type(16)));
typedef float    v8f   __attribute__((ext_vector_type(8)));

#define B_  1024
#define T_  512
#define DIN 4
#define H_  64
#define G_  256

#if __has_builtin(__builtin_amdgcn_rcpf)
#define RCPF(x) __builtin_amdgcn_rcpf(x)
#else
#define RCPF(x) (1.0f / (x))
#endif
#if __has_builtin(__builtin_amdgcn_exp2f)
#define EXP2F(x) __builtin_amdgcn_exp2f(x)
#else
#define EXP2F(x) exp2f(x)
#endif
#define LOG2E 1.4426950408889634f

__device__ __forceinline__ float fast_tanh(float x) {
#if __has_builtin(__builtin_amdgcn_tanhf)
    return __builtin_amdgcn_tanhf(x);            // V_TANH_F32 (CDNA5 TRANS op)
#elif __has_builtin(__builtin_amdgcn_tanh_f32)
    return __builtin_amdgcn_tanh_f32(x);
#else
    return 1.0f - 2.0f * RCPF(1.0f + EXP2F(x * (2.0f * LOG2E)));
#endif
}
// sigmoid(x) = 0.5*tanh(0.5x) + 0.5  -> mul, v_tanh, fma (branch-free)
__device__ __forceinline__ float fast_sig(float x) {
    return __builtin_fmaf(fast_tanh(x * 0.5f), 0.5f, 0.5f);
}

__device__ __forceinline__ v16h cat8(half8 lo, half8 hi) {
    return __builtin_shufflevector(lo, hi, 0,1,2,3,4,5,6,7,8,9,10,11,12,13,14,15);
}

// ---------------------------------------------------------------------------
// Pack fp32 weights into f16 WMMA B-fragments.
// B layout: slot j of lane: k = 32*q + ((lane>=16)?16:0) + j, col = lane&15.
// Tile n (0..15): gt = n>>2 (gate type), w = n&3 (h-col slice) ->
//   gate index g = 64*gt + 16*w + col.
// layer0: 2 kchunks x 16 tiles = 32 frags; layers1-3: 4 x 16 = 64 each.
// frag bases: {0, 32, 96, 160}; 224 frags, 512 f16 (1 KB) each.
// ---------------------------------------------------------------------------
__global__ void pack_weights(const float* __restrict__ whh0,
                             const float* __restrict__ wih1, const float* __restrict__ whh1,
                             const float* __restrict__ wih2, const float* __restrict__ whh2,
                             const float* __restrict__ wih3, const float* __restrict__ whh3,
                             _Float16* __restrict__ wfrag) {
    int frag = blockIdx.x;          // 0..223
    int lane = threadIdx.x;         // 0..31
    int l, rem;
    if (frag < 32) { l = 0; rem = frag; }
    else { int f2 = frag - 32; l = 1 + f2 / 64; rem = f2 % 64; }
    int q = rem >> 4;
    int n = rem & 15;
    int g = 64 * (n >> 2) + 16 * (n & 3) + (lane & 15);   // permuted tile->gate map

    const float* wih = (l == 1) ? wih1 : (l == 2) ? wih2 : wih3;
    const float* whh = (l == 0) ? whh0 : (l == 1) ? whh1 : (l == 2) ? whh2 : whh3;

    half8 o0, o1;
    #pragma unroll
    for (int j = 0; j < 16; ++j) {
        int k = 32 * q + ((lane >> 4) << 4) + j;   // B-matrix K index
        float v;
        if (l == 0) v = whh[g * 64 + k];
        else        v = (k < 64) ? wih[g * 64 + k] : whh[g * 64 + (k - 64)];
        if (j < 8) o0[j] = (_Float16)v; else o1[j - 8] = (_Float16)v;
    }
    half8* op = (half8*)(wfrag + (size_t)frag * 512 + lane * 16);
    op[0] = o0;
    op[1] = o1;
}

// ---------------------------------------------------------------------------
// Fully fused 4-layer LSTM scan + MLP head + softmax.
// grid = 64 (batch tiles of 16), block = 128 (4 waves).
// Wave w owns h-columns [16w,16w+16): 4 independent WMMA chains (i/f/g/o),
// in-register cell state, f16 h to double-buffered LDS, 1 barrier per layer.
// x(t+1) is prefetched into registers and staged into LDS under layer 3's
// barrier, so the scan runs on exactly 4 barriers per time step.
// ---------------------------------------------------------------------------
__launch_bounds__(128, 1)
__global__ void lstm_fused(const float* __restrict__ x,
                           const float* __restrict__ wih0,
                           const float* __restrict__ bih0, const float* __restrict__ bhh0,
                           const float* __restrict__ bih1, const float* __restrict__ bhh1,
                           const float* __restrict__ bih2, const float* __restrict__ bhh2,
                           const float* __restrict__ bih3, const float* __restrict__ bhh3,
                           const float* __restrict__ fc1w, const float* __restrict__ fc1b,
                           const float* __restrict__ fc2w, const float* __restrict__ fc2b,
                           const _Float16* __restrict__ wfrag,
                           float* __restrict__ out) {
    __shared__ __attribute__((aligned(16))) _Float16 h_state[4][2][16][64]; // [layer][slot]
    __shared__ float biasL[4][256];
    __shared__ float w0s[256 * 4];
    __shared__ float xbuf[16][4];
    __shared__ float zbuf[16 * 32];
    __shared__ float lgbuf[16 * 3];

    const int tid  = threadIdx.x;
    const int lane = tid & 31;
    const int wid  = tid >> 5;            // 0..3 -> h-col slice [16w,16w+16)
    const int base = blockIdx.x * 16;     // batch tile

    // ---- one-time init (also fills xbuf for t=0) ----
    for (int i = tid; i < 4 * 2 * 16 * 64; i += 128)
        ((_Float16*)h_state)[i] = (_Float16)0.0f;
    for (int i = tid; i < 1024; i += 128) {
        w0s[i] = wih0[i];
        int l = i >> 8, g = i & 255;
        const float* bi = (l == 0) ? bih0 : (l == 1) ? bih1 : (l == 2) ? bih2 : bih3;
        const float* bh = (l == 0) ? bhh0 : (l == 1) ? bhh1 : (l == 2) ? bhh2 : bhh3;
        ((float*)biasL)[i] = bi[g] + bh[g];
    }
    if (tid < 64)
        xbuf[tid >> 2][tid & 3] = x[((size_t)(base + (tid >> 2)) * T_ + 0) * DIN + (tid & 3)];
    __syncthreads();

    const int nloc  = lane & 15;
    const int rbase = (lane >> 4) << 3;      // C/D rows: 0..7 or 8..15
    const int arow  = nloc;                  // A row for this lane
    const int ksel  = (lane >> 4) << 4;      // A K split: +0 / +16
    const int hcol  = 16 * wid + nloc;       // this lane's h column

    int gidx[4];
    #pragma unroll
    for (int gt = 0; gt < 4; ++gt) gidx[gt] = 64 * gt + hcol;

    float creg[4][8];                        // cell state, registers only
    #pragma unroll
    for (int l = 0; l < 4; ++l)
        #pragma unroll
        for (int j = 0; j < 8; ++j) creg[l][j] = 0.0f;

    for (int t = 0; t < T_; ++t) {
        const int ws = t & 1;                // write slot (h_t)
        const int rs = 1 - ws;               // read slot  (h_{t-1})

        // prefetch x(t+1) into registers; latency hidden behind this step
        float xr = 0.0f;
        if (tid < 64 && t + 1 < T_)
            xr = x[((size_t)(base + (tid >> 2)) * T_ + (t + 1)) * DIN + (tid & 3)];

        // ---------------- layer 0: x-part via VALU (K=4), h-part via WMMA ----
        {
            v8f acc[4];
            #pragma unroll
            for (int gt = 0; gt < 4; ++gt) {
                float bv  = biasL[0][gidx[gt]];
                float w00 = w0s[gidx[gt] * 4 + 0], w01 = w0s[gidx[gt] * 4 + 1];
                float w02 = w0s[gidx[gt] * 4 + 2], w03 = w0s[gidx[gt] * 4 + 3];
                #pragma unroll
                for (int j = 0; j < 8; ++j) {
                    int r = rbase + j;
                    acc[gt][j] = bv + xbuf[r][0] * w00 + xbuf[r][1] * w01
                                    + xbuf[r][2] * w02 + xbuf[r][3] * w03;
                }
            }
            v16h a[2];                              // all A frags up front
            #pragma unroll
            for (int q = 0; q < 2; ++q) {
                const half8* ap = (const half8*)&h_state[0][rs][arow][q * 32 + ksel];
                a[q] = cat8(ap[0], ap[1]);
            }
            #pragma unroll
            for (int q = 0; q < 2; ++q)
                #pragma unroll
                for (int gt = 0; gt < 4; ++gt) {
                    const half8* bp = (const half8*)(wfrag +
                        (size_t)(q * 16 + gt * 4 + wid) * 512 + lane * 16);
                    v16h b = cat8(bp[0], bp[1]);
                    acc[gt] = __builtin_amdgcn_wmma_f32_16x16x32_f16(false, a[q], false, b,
                                                                     (short)0, acc[gt], false, false);
                }
            #pragma unroll
            for (int j = 0; j < 8; ++j) {
                float iv = fast_sig(acc[0][j]);
                float fv = fast_sig(acc[1][j]);
                float gv = fast_tanh(acc[2][j]);
                float ov = fast_sig(acc[3][j]);
                float c  = fv * creg[0][j] + iv * gv;
                creg[0][j] = c;
                h_state[0][ws][rbase + j][hcol] = (_Float16)(ov * fast_tanh(c));
            }
        }
        __syncthreads();

        // ---------------- layers 1..3: K=128 ([h_{l-1,t} | h_{l,t-1}]) ------
        #pragma unroll
        for (int l = 1; l < 4; ++l) {
            const int fbase = (l == 1) ? 32 : (l == 2) ? 96 : 160;
            v8f acc[4];
            #pragma unroll
            for (int gt = 0; gt < 4; ++gt) {
                float bv = biasL[l][gidx[gt]];
                #pragma unroll
                for (int j = 0; j < 8; ++j) acc[gt][j] = bv;
            }
            v16h a[4];                              // all A frags up front
            #pragma unroll
            for (int q = 0; q < 4; ++q) {
                const _Float16* hs = (q < 2) ? &h_state[l - 1][ws][0][0]
                                             : &h_state[l][rs][0][0];
                const half8* ap = (const half8*)(hs + arow * 64 + (q & 1) * 32 + ksel);
                a[q] = cat8(ap[0], ap[1]);
            }
            #pragma unroll
            for (int q = 0; q < 4; ++q)
                #pragma unroll
                for (int gt = 0; gt < 4; ++gt) {
                    const half8* bp = (const half8*)(wfrag +
                        (size_t)(fbase + q * 16 + gt * 4 + wid) * 512 + lane * 16);
                    v16h b = cat8(bp[0], bp[1]);
                    acc[gt] = __builtin_amdgcn_wmma_f32_16x16x32_f16(false, a[q], false, b,
                                                                     (short)0, acc[gt], false, false);
                }
            #pragma unroll
            for (int j = 0; j < 8; ++j) {
                float iv = fast_sig(acc[0][j]);
                float fv = fast_sig(acc[1][j]);
                float gv = fast_tanh(acc[2][j]);
                float ov = fast_sig(acc[3][j]);
                float c  = fv * creg[l][j] + iv * gv;
                creg[l][j] = c;
                h_state[l][ws][rbase + j][hcol] = (_Float16)(ov * fast_tanh(c));
            }
            if (l == 3 && tid < 64)                // stage x(t+1) under layer-3 barrier
                xbuf[tid >> 2][tid & 3] = xr;
            __syncthreads();
        }
    }

    // ---------------- head: fc1+relu, fc2, softmax ---------------------------
    const int fs = (T_ - 1) & 1;            // final slot of h_state[3]
    for (int o = tid; o < 16 * 32; o += 128) {
        int r = o >> 5, m = o & 31;
        float acc = fc1b[m];
        #pragma unroll
        for (int k = 0; k < 64; ++k)
            acc += (float)h_state[3][fs][r][k] * fc1w[m * 64 + k];
        zbuf[o] = fmaxf(acc, 0.0f);
    }
    __syncthreads();
    if (tid < 48) {
        int r = tid / 3, c = tid % 3;
        float acc = fc2b[c];
        #pragma unroll
        for (int m = 0; m < 32; ++m) acc += zbuf[r * 32 + m] * fc2w[c * 32 + m];
        lgbuf[r * 3 + c] = acc;
    }
    __syncthreads();
    if (tid < 16) {
        float a0 = lgbuf[tid * 3 + 0], a1 = lgbuf[tid * 3 + 1], a2 = lgbuf[tid * 3 + 2];
        float mx = fmaxf(a0, fmaxf(a1, a2));
        float e0 = EXP2F((a0 - mx) * LOG2E);
        float e1 = EXP2F((a1 - mx) * LOG2E);
        float e2 = EXP2F((a2 - mx) * LOG2E);
        float s = RCPF(e0 + e1 + e2);
        size_t o = (size_t)(base + tid) * 3;
        out[o + 0] = e0 * s;
        out[o + 1] = e1 * s;
        out[o + 2] = e2 * s;
    }
}

extern "C" void kernel_launch(void* const* d_in, const int* in_sizes, int n_in,
                              void* d_out, int out_size, void* d_ws, size_t ws_size,
                              hipStream_t stream) {
    const float* x    = (const float*)d_in[0];
    const float* wih0 = (const float*)d_in[1];
    const float* whh0 = (const float*)d_in[2];
    const float* bih0 = (const float*)d_in[3];
    const float* bhh0 = (const float*)d_in[4];
    const float* wih1 = (const float*)d_in[5];
    const float* whh1 = (const float*)d_in[6];
    const float* bih1 = (const float*)d_in[7];
    const float* bhh1 = (const float*)d_in[8];
    const float* wih2 = (const float*)d_in[9];
    const float* whh2 = (const float*)d_in[10];
    const float* bih2 = (const float*)d_in[11];
    const float* bhh2 = (const float*)d_in[12];
    const float* wih3 = (const float*)d_in[13];
    const float* whh3 = (const float*)d_in[14];
    const float* bih3 = (const float*)d_in[15];
    const float* bhh3 = (const float*)d_in[16];
    const float* fc1w = (const float*)d_in[17];
    const float* fc1b = (const float*)d_in[18];
    const float* fc2w = (const float*)d_in[19];
    const float* fc2b = (const float*)d_in[20];

    _Float16* wfrag = (_Float16*)d_ws;   // 224 KB of packed B-fragments

    pack_weights<<<224, 32, 0, stream>>>(whh0, wih1, whh1, wih2, whh2, wih3, whh3, wfrag);

    lstm_fused<<<B_ / 16, 128, 0, stream>>>(x, wih0,
                                            bih0, bhh0, bih1, bhh1, bih2, bhh2, bih3, bhh3,
                                            fc1w, fc1b, fc2w, fc2b,
                                            wfrag, (float*)d_out);
}
